// VariationalGCNEncoder_9783935500965
// MI455X (gfx1250) — compile-verified, matrix operand order checked by reading, not run on hardware
//
#include <hip/hip_runtime.h>
#include <hip/hip_bf16.h>

typedef __attribute__((ext_vector_type(16))) _Float16 v16h;
typedef __attribute__((ext_vector_type(8)))  float    v8f;

// ---------------- degree / normalization ----------------

__global__ void k_init_deg(float* __restrict__ deg, int n) {
    int i = blockIdx.x * blockDim.x + threadIdx.x;
    if (i < n) deg[i] = 1.0f;   // self-loop contributes 1
}

__global__ void k_deg_edges(const int* __restrict__ col, float* __restrict__ deg, int E) {
    int e = blockIdx.x * blockDim.x + threadIdx.x;
    if (e < E) atomicAdd(&deg[col[e]], 1.0f);
}

__global__ void k_rsqrt_inplace(float* __restrict__ d, int n) {
    int i = blockIdx.x * blockDim.x + threadIdx.x;
    if (i < n) d[i] = rsqrtf(d[i]);   // deg >= 1 always (self-loop)
}

// ---------------- WMMA GEMM: Out[M,N] = (relu?)A[M,K] * B[K,N] ----------------
// One wave computes one 16x16 tile. f32 -> f16 convert on load, f32 accumulate.
// A-fragment layout (16-bit 16x32): lanes 0-15 M=lane (kbase 0), lanes 16-31
// M=lane-16 (kbase 8); VGPR i<4 holds K=kbase+2i..+1, VGPR 4+i holds K=16+kbase+2i..+1.
// B mirrors A with N in place of M. C/D: VGPR r -> M = r + (lane>=16 ? 8 : 0), N = lane&15.
// K and RELU are compile-time: full unroll -> 4 back-to-back v_wmma, no inner branches.

template <int K, int RELU>
__global__ __launch_bounds__(256)
void k_gemm_wmma_f16(const float* __restrict__ A, const float* __restrict__ B,
                     float* __restrict__ Out, int M, int N)
{
    const int lane  = threadIdx.x & 31;
    const int wave  = blockIdx.x * (blockDim.x >> 5) + (threadIdx.x >> 5);
    const int ntile = N >> 4;
    const int mt = wave / ntile;
    const int nt = wave % ntile;
    if (mt >= (M >> 4)) return;           // wave-uniform: EXEC stays all-ones

    const int l15   = lane & 15;
    const int kbase = (lane & 16) ? 8 : 0;
    const int mrow  = (mt << 4) + l15;
    const int ncol  = (nt << 4) + l15;
    const float* __restrict__ arow = A + (size_t)mrow * K;

    v8f acc = {0.f, 0.f, 0.f, 0.f, 0.f, 0.f, 0.f, 0.f};

#pragma unroll
    for (int kc = 0; kc < K; kc += 32) {
        v16h af, bf;
#pragma unroll
        for (int i = 0; i < 4; ++i) {
            const int k0 = kc + kbase + 2 * i;   // even -> 8B aligned
            const int k1 = k0 + 16;
            float2 p0 = *(const float2*)(arow + k0);
            float2 p1 = *(const float2*)(arow + k1);
            if (RELU) {
                p0.x = fmaxf(p0.x, 0.f); p0.y = fmaxf(p0.y, 0.f);
                p1.x = fmaxf(p1.x, 0.f); p1.y = fmaxf(p1.y, 0.f);
            }
            af[2 * i]     = (_Float16)p0.x;
            af[2 * i + 1] = (_Float16)p0.y;
            af[8 + 2 * i] = (_Float16)p1.x;
            af[9 + 2 * i] = (_Float16)p1.y;
            bf[2 * i]     = (_Float16)B[(size_t)k0 * N + ncol];
            bf[2 * i + 1] = (_Float16)B[(size_t)(k0 + 1) * N + ncol];
            bf[8 + 2 * i] = (_Float16)B[(size_t)k1 * N + ncol];
            bf[9 + 2 * i] = (_Float16)B[(size_t)(k1 + 1) * N + ncol];
        }
        acc = __builtin_amdgcn_wmma_f32_16x16x32_f16(
            /*neg_a=*/false, af, /*neg_b=*/false, bf,
            /*c_mod=*/(short)0, acc, /*reuse_a=*/false, /*reuse_b=*/false);
    }

    const int mb = (mt << 4) + ((lane & 16) ? 8 : 0);
#pragma unroll
    for (int r = 0; r < 8; ++r)
        Out[(size_t)(mb + r) * N + ncol] = acc[r];
}

// ---------------- aggregation: Out = bias + dinv^2 * S (self loop), then edge scatter ----------------

__global__ void k_agg_init(const float* __restrict__ S, const float* __restrict__ dinv,
                           const float* __restrict__ bias, float* __restrict__ Out,
                           int n, int F)
{
    long long idx = (long long)blockIdx.x * blockDim.x + threadIdx.x;
    long long total = (long long)n * F;
    if (idx >= total) return;
    int i = (int)(idx / F);
    int f = (int)(idx - (long long)i * F);
    float di = dinv[i];
    Out[idx] = bias[f] + di * di * S[idx];
}

// One thread = one (edge, 4-feature) group. qshift = log2(F/4); F in {128,64}.
// With F=128 a warp covers exactly one edge: row/col/dinv loads are warp-uniform
// broadcasts and the float4 gather is a fully coalesced 512B read.
__global__ void k_agg_edges(const int* __restrict__ row, const int* __restrict__ col,
                            const float* __restrict__ S, const float* __restrict__ dinv,
                            float* __restrict__ Out, long long E, int F, int qshift)
{
    long long idx = (long long)blockIdx.x * blockDim.x + threadIdx.x;
    long long total = E << qshift;
    if (idx >= total) return;
    long long e = idx >> qshift;
    int q = (int)(idx & ((1 << qshift) - 1));
    int r = row[e];
    int c = col[e];
    float nrm = dinv[r] * dinv[c];
    const float4 s = ((const float4*)(S + (size_t)r * F))[q];
    float* dst = Out + (size_t)c * F + (q << 2);
    atomicAdd(dst + 0, nrm * s.x);
    atomicAdd(dst + 1, nrm * s.y);
    atomicAdd(dst + 2, nrm * s.z);
    atomicAdd(dst + 3, nrm * s.w);
}

// ---------------- launch ----------------

extern "C" void kernel_launch(void* const* d_in, const int* in_sizes, int n_in,
                              void* d_out, int out_size, void* d_ws, size_t ws_size,
                              hipStream_t stream) {
    const float* x   = (const float*)d_in[0];
    const int*   ei  = (const int*)d_in[1];
    const float* W1  = (const float*)d_in[2];
    const float* b1  = (const float*)d_in[3];
    const float* Wmu = (const float*)d_in[4];
    const float* bmu = (const float*)d_in[5];
    const float* Wls = (const float*)d_in[6];
    const float* bls = (const float*)d_in[7];

    const int F = 128, Hd = 128, Z = 64;
    const int n = in_sizes[0] / F;                 // 50000
    const long long E = in_sizes[1] / 2;           // 1.6M
    const int*   row = ei;
    const int*   col = ei + E;

    float* out_mu = (float*)d_out;
    float* out_ls = out_mu + (size_t)n * Z;

    // workspace carve-up (floats): dinv | XW1 | Hbuf | HWmu | HWls  (~77 MB)
    float* ws   = (float*)d_ws;
    float* dinv = ws;
    float* XW1  = ws + (((size_t)n + 255) & ~(size_t)255);
    float* Hbuf = XW1  + (size_t)n * Hd;
    float* HWmu = Hbuf + (size_t)n * Hd;
    float* HWls = HWmu + (size_t)n * Z;

    const int TB = 256;
    dim3 blk(TB);

    // 1) degree -> dinv (shared by all three convs)
    k_init_deg     <<<(n + TB - 1) / TB, blk, 0, stream>>>(dinv, n);
    k_deg_edges    <<<(int)((E + TB - 1) / TB), blk, 0, stream>>>(col, dinv, (int)E);
    k_rsqrt_inplace<<<(n + TB - 1) / TB, blk, 0, stream>>>(dinv, n);

    // 2) conv1: XW1 = x @ W1 ; H = b1 + norm-weighted scatter ; (relu fused into next GEMMs)
    {
        int tiles = (n >> 4) * (Hd >> 4);
        k_gemm_wmma_f16<128, 0><<<(tiles + 7) / 8, blk, 0, stream>>>(x, W1, XW1, n, Hd);
    }
    {
        long long tot = (long long)n * Hd;
        k_agg_init<<<(int)((tot + TB - 1) / TB), blk, 0, stream>>>(XW1, dinv, b1, Hbuf, n, Hd);
        long long etot = E << 5;  // F=128 -> 32 float4 groups
        k_agg_edges<<<(int)((etot + TB - 1) / TB), blk, 0, stream>>>(row, col, XW1, dinv, Hbuf, E, Hd, 5);
    }

    // 3) conv_mu / conv_ls dense transforms: relu(H) @ W  (relu fused into A-load)
    {
        int tiles = (n >> 4) * (Z >> 4);
        k_gemm_wmma_f16<128, 1><<<(tiles + 7) / 8, blk, 0, stream>>>(Hbuf, Wmu, HWmu, n, Z);
        k_gemm_wmma_f16<128, 1><<<(tiles + 7) / 8, blk, 0, stream>>>(Hbuf, Wls, HWls, n, Z);
    }

    // 4) aggregate straight into d_out
    {
        long long tot = (long long)n * Z;
        long long etot = E << 4;  // F=64 -> 16 float4 groups
        k_agg_init<<<(int)((tot + TB - 1) / TB), blk, 0, stream>>>(HWmu, dinv, bmu, out_mu, n, Z);
        k_agg_edges<<<(int)((etot + TB - 1) / TB), blk, 0, stream>>>(row, col, HWmu, dinv, out_mu, E, Z, 4);
        k_agg_init<<<(int)((tot + TB - 1) / TB), blk, 0, stream>>>(HWls, dinv, bls, out_ls, n, Z);
        k_agg_edges<<<(int)((etot + TB - 1) / TB), blk, 0, stream>>>(row, col, HWls, dinv, out_ls, E, Z, 4);
    }
}
